// MultiHeadAttention_3427383902885
// MI455X (gfx1250) — compile-verified, matrix operand order checked by reading, not run on hardware
//
#include <hip/hip_runtime.h>

#define DEVINL __device__ __forceinline__

typedef __attribute__((ext_vector_type(16))) __bf16 v16bf;
typedef __attribute__((ext_vector_type(8)))  float  v8f;
typedef __attribute__((ext_vector_type(4)))  int    v4i;

#if defined(__AMDGCN__) && __has_builtin(__builtin_amdgcn_global_load_async_to_lds_b128)
#define HAVE_ASYNC_LDS 1
#else
#define HAVE_ASYNC_LDS 0
#endif

DEVINL void wait_asynccnt0() {
#if defined(__AMDGCN__)
#if __has_builtin(__builtin_amdgcn_s_wait_asynccnt)
  __builtin_amdgcn_s_wait_asynccnt(0);
#else
  asm volatile("s_wait_asynccnt 0" ::: "memory");
#endif
#endif
}

// ---------------- problem constants ----------------
constexpr int B = 32, S = 1024, D = 192, H = 8, DH = 24, DHP = 32;
constexpr int NQKV = H * 3 * DHP;        // 768 combined (padded) QKV columns
constexpr int KT   = D / 32;             // 6 K-steps of 32
constexpr int NT_QKV = NQKV / 16;        // 48 N-tiles
constexpr int NT_O   = D / 16;           // 12 N-tiles
constexpr int MROWS  = B * S;            // 32768
constexpr float SCALE = 0.20412414523193154f; // 1/sqrt(24)

// ---------------- workspace layout (bytes) ----------------
constexpr size_t OFF_WB_QKV  = 0;
constexpr size_t SZ_WB_QKV   = (size_t)NT_QKV * KT * 32 * 16 * 2;        // 294912
constexpr size_t OFF_B_QKV   = OFF_WB_QKV + SZ_WB_QKV;
constexpr size_t SZ_B_QKV    = (size_t)NQKV * 4;
constexpr size_t OFF_WB_O    = ((OFF_B_QKV + SZ_B_QKV + 255) / 256) * 256;
constexpr size_t SZ_WB_O     = (size_t)NT_O * KT * 32 * 16 * 2;          // 73728
constexpr size_t OFF_B_O     = OFF_WB_O + SZ_WB_O;
constexpr size_t SZ_B_O      = (size_t)D * 4;
constexpr size_t OFF_QKV     = ((OFF_B_O + SZ_B_O + 255) / 256) * 256;
constexpr size_t SZ_QKV      = (size_t)MROWS * NQKV * 2;                 // ~50.3 MB bf16
constexpr size_t OFF_ATTO    = ((OFF_QKV + SZ_QKV + 255) / 256) * 256;   // fp32 O, 25.2 MB

// A/B fragment K-index mapping for 16-bit 16x32 / 32x16 tiles (wave32):
// lanes 0-15 hold K {0..7,16..23}; lanes 16-31 hold K {8..15,24..31}
DEVINL int kmap(int lane, int i) { return ((i < 8) ? i : i + 8) + ((lane < 16) ? 0 : 8); }

union FragU { uint4 u[2]; v16bf v; };

DEVINL float rmax16(float v) {
#pragma unroll
  for (int o = 8; o; o >>= 1) v = fmaxf(v, __shfl_xor(v, o, 16));
  return v;
}
DEVINL float rsum16(float v) {
#pragma unroll
  for (int o = 8; o; o >>= 1) v += __shfl_xor(v, o, 16);
  return v;
}

// ---------------- weight prep: QKV combined, padded, B-fragment layout ----------------
__global__ void prep_qkv_w(const float* __restrict__ Wq, const float* __restrict__ bq,
                           const float* __restrict__ Wk, const float* __restrict__ bk,
                           const float* __restrict__ Wv, const float* __restrict__ bv,
                           __bf16* __restrict__ wb, float* __restrict__ bias) {
  int id = blockIdx.x * 256 + threadIdx.x;
  constexpr int TOT = NT_QKV * KT * 32 * 16;
  if (id < TOT) {
    int i = id & 15, lane = (id >> 4) & 31;
    int kt = (id >> 9) % KT, nT = (id >> 9) / KT;
    int n = nT * 16 + (lane & 15);
    int k = kt * 32 + kmap(lane, i);
    int h = n / 96, r = n % 96, sel = r / 32, e = r & 31;
    const float* W = (sel == 0) ? Wq : ((sel == 1) ? Wk : Wv);
    float v = (e < DH) ? W[(h * D + k) * DH + e] : 0.0f;
    wb[id] = (__bf16)v;
  }
  if (id < NQKV) {
    int h = id / 96, r = id % 96, sel = r / 32, e = r & 31;
    const float* bb = (sel == 0) ? bq : ((sel == 1) ? bk : bv);
    bias[id] = (e < DH) ? bb[h * DH + e] : 0.0f;
  }
}

__global__ void prep_o_w(const float* __restrict__ Wo, const float* __restrict__ bo,
                         __bf16* __restrict__ wb, float* __restrict__ bias) {
  int id = blockIdx.x * 256 + threadIdx.x;
  constexpr int TOT = NT_O * KT * 32 * 16;
  if (id < TOT) {
    int i = id & 15, lane = (id >> 4) & 31;
    int kt = (id >> 9) % KT, nT = (id >> 9) / KT;
    int n = nT * 16 + (lane & 15);
    int k = kt * 32 + kmap(lane, i);
    wb[id] = (__bf16)Wo[k * D + n];
  }
  if (id < D) bias[id] = bo[id];
}

// ---------------- WMMA GEMM: [M x 192] f32  x  [192 x N] bf16-frag  -> out (+bias) ----
// MODE 0: f32 out, MODE 1: bf16 out. Block: 256 thr = 8 waves; tile 64(M) x 64(N).
template <int MODE>
__global__ void gemm_wmma(const float* __restrict__ X, const __bf16* __restrict__ WB,
                          const float* __restrict__ bias, void* __restrict__ out,
                          int ldOut) {
  __shared__ __bf16 xs[64 * 32];  // 4 KB staged A tile (bf16)
  const int tid = threadIdx.x;
  const int lane = tid & 31, w = tid >> 5;
  const int wm = w & 3, wn = w >> 2;
  const int m0 = blockIdx.x * 64;
  const int nt0 = blockIdx.y * 4 + wn * 2;

  const v16bf* wbv = (const v16bf*)WB;
  v8f c0 = {0.f, 0.f, 0.f, 0.f, 0.f, 0.f, 0.f, 0.f};
  v8f c1 = c0;

  const int srow = tid >> 2;            // 0..63
  const int skb  = (tid & 3) * 8;       // 0,8,16,24

#pragma unroll
  for (int kt = 0; kt < KT; ++kt) {
    // stage X[64 x 32] f32 -> bf16 LDS
    const float* xp = X + (size_t)(m0 + srow) * D + kt * 32 + skb;
    float4 f0 = *(const float4*)(xp);
    float4 f1 = *(const float4*)(xp + 4);
    __bf16* xd = &xs[srow * 32 + skb];
    xd[0] = (__bf16)f0.x; xd[1] = (__bf16)f0.y; xd[2] = (__bf16)f0.z; xd[3] = (__bf16)f0.w;
    xd[4] = (__bf16)f1.x; xd[5] = (__bf16)f1.y; xd[6] = (__bf16)f1.z; xd[7] = (__bf16)f1.w;
    __syncthreads();

    // A fragment from LDS
    const int arow = wm * 16 + (lane & 15);
    const int base = (lane < 16) ? 0 : 8;
    FragU a;
    a.u[0] = *(const uint4*)&xs[arow * 32 + base];
    a.u[1] = *(const uint4*)&xs[arow * 32 + base + 16];

    // B fragments: contiguous v16bf per lane
    v16bf b0 = wbv[(size_t)(nt0 * KT + kt) * 32 + lane];
    v16bf b1 = wbv[(size_t)((nt0 + 1) * KT + kt) * 32 + lane];

    c0 = __builtin_amdgcn_wmma_f32_16x16x32_bf16(false, a.v, false, b0, (short)0, c0, false, false);
    c1 = __builtin_amdgcn_wmma_f32_16x16x32_bf16(false, a.v, false, b1, (short)0, c1, false, false);
    __syncthreads();
  }

  // epilogue: C layout -> row = r (+8 for upper half-wave), col = lane%16
  const int colbase = blockIdx.y * 64 + wn * 32;
  const int rhalf = (lane < 16) ? 0 : 8;
#pragma unroll
  for (int f = 0; f < 2; ++f) {
    v8f c = f ? c1 : c0;
    int col = colbase + f * 16 + (lane & 15);
    float bia = bias[col];
#pragma unroll
    for (int r = 0; r < 8; ++r) {
      int row = m0 + wm * 16 + r + rhalf;
      float v = c[r] + bia;
      if (MODE == 0) ((float*)out)[(size_t)row * ldOut + col] = v;
      else           ((__bf16*)out)[(size_t)row * ldOut + col] = (__bf16)v;
    }
  }
}

// ---------------- flash attention: block = 4 waves = 64 query rows of one (b,h) -------
__global__ void flash_attn(const __bf16* __restrict__ qkv,
                           const unsigned char* __restrict__ mask,
                           float* __restrict__ O) {
  __shared__ __bf16 kraw[32 * 32];       // K chunk, row = key, col = feature
  __shared__ __bf16 vT[32 * 32];         // V chunk transposed: row = feature, col = key
  __shared__ __bf16 pbuf[4 * 16 * 32];   // per-wave P transpose buffer

  const int qb = blockIdx.x & 15;
  const int h  = (blockIdx.x >> 4) & 7;
  const int b  = blockIdx.x >> 7;
  const int tid = threadIdx.x, lane = tid & 31, w = tid >> 5;
  const int base = (lane < 16) ? 0 : 8;
  const int rhalf = (lane < 16) ? 0 : 8;

  // Q A-fragment (held in registers for the whole key loop)
  const int qrow = qb * 64 + w * 16 + (lane & 15);
  const __bf16* qp = qkv + (size_t)(b * S + qrow) * NQKV + h * 96;
  FragU aq;
  aq.u[0] = *(const uint4*)(qp + base);
  aq.u[1] = *(const uint4*)(qp + base + 16);

  v8f acc0 = {0.f, 0.f, 0.f, 0.f, 0.f, 0.f, 0.f, 0.f};
  v8f acc1 = acc0;
  float m[8], l[8];
#pragma unroll
  for (int r = 0; r < 8; ++r) { m[r] = -3.0e38f; l[r] = 0.0f; }

  const int skey = tid >> 2;            // 0..31 (staging key)
  const int sfo  = (tid & 3) * 8;       // 0,8,16,24 (staging feature base)

  for (int j = 0; j < S / 32; ++j) {
    const int t0 = j * 32;
    // stage K chunk (row-major) and V chunk (transposed) into LDS
    const __bf16* kg = qkv + (size_t)(b * S + t0 + skey) * NQKV + h * 96 + 32 + sfo;
    const __bf16* vg = qkv + (size_t)(b * S + t0 + skey) * NQKV + h * 96 + 64 + sfo;

#if HAVE_ASYNC_LDS
    // CDNA5 async copy: global -> LDS without touching VGPRs (ASYNCcnt-tracked)
    __builtin_amdgcn_global_load_async_to_lds_b128((v4i*)kg, (v4i*)&kraw[skey * 32 + sfo], 0, 0);
#else
    {
      uint4 kq = *(const uint4*)kg;
      *(uint4*)&kraw[skey * 32 + sfo] = kq;
    }
#endif
    // V is transposed through registers (feature-major LDS layout for PV B-fragments)
    uint4 vq = *(const uint4*)vg;
    const __bf16* vh = (const __bf16*)&vq;
#pragma unroll
    for (int i = 0; i < 8; ++i) vT[(sfo + i) * 32 + skey] = vh[i];
    if (j + 1 < S / 32) __builtin_prefetch(kg + NQKV * 32, 0, 1);
#if HAVE_ASYNC_LDS
    wait_asynccnt0();
#endif
    __syncthreads();

    // scores: S = Q * K^T  (two 16-key groups)
    FragU bk0, bk1;
    const int keyl = lane & 15;
    bk0.u[0] = *(const uint4*)&kraw[keyl * 32 + base];
    bk0.u[1] = *(const uint4*)&kraw[keyl * 32 + base + 16];
    bk1.u[0] = *(const uint4*)&kraw[(keyl + 16) * 32 + base];
    bk1.u[1] = *(const uint4*)&kraw[(keyl + 16) * 32 + base + 16];
    v8f z = {0.f, 0.f, 0.f, 0.f, 0.f, 0.f, 0.f, 0.f};
    v8f s0 = __builtin_amdgcn_wmma_f32_16x16x32_bf16(false, aq.v, false, bk0.v, (short)0, z, false, false);
    v8f s1 = __builtin_amdgcn_wmma_f32_16x16x32_bf16(false, aq.v, false, bk1.v, (short)0, z, false, false);

    // scale + mask (mask indexed by key position)
    const bool mk0 = mask[b * S + t0 + keyl] != 0;
    const bool mk1 = mask[b * S + t0 + 16 + keyl] != 0;
#pragma unroll
    for (int r = 0; r < 8; ++r) {
      s0[r] = mk0 ? s0[r] * SCALE : -1.0e9f;
      s1[r] = mk1 ? s1[r] * SCALE : -1.0e9f;
    }

    // online softmax + P write (C layout -> A layout via per-wave LDS)
#pragma unroll
    for (int r = 0; r < 8; ++r) {
      float rm = rmax16(fmaxf(s0[r], s1[r]));
      float mn = fmaxf(m[r], rm);
      float corr = __expf(m[r] - mn);
      float p0 = __expf(s0[r] - mn);
      float p1 = __expf(s1[r] - mn);
      l[r] = l[r] * corr + rsum16(p0 + p1);
      m[r] = mn;
      acc0[r] *= corr;
      acc1[r] *= corr;
      int prow = r + rhalf;
      pbuf[w * 512 + prow * 32 + keyl]      = (__bf16)p0;
      pbuf[w * 512 + prow * 32 + 16 + keyl] = (__bf16)p1;
    }
    // ensure all P stores of this wave have landed before re-reading (in-order LDS + hw wait)
    asm volatile("s_wait_dscnt 0" ::: "memory");

    FragU ap;
    ap.u[0] = *(const uint4*)&pbuf[w * 512 + (lane & 15) * 32 + base];
    ap.u[1] = *(const uint4*)&pbuf[w * 512 + (lane & 15) * 32 + base + 16];

    // O += P * V   (V^T layout in LDS makes B fragments contiguous)
    FragU bv0, bv1;
    bv0.u[0] = *(const uint4*)&vT[(lane & 15) * 32 + base];
    bv0.u[1] = *(const uint4*)&vT[(lane & 15) * 32 + base + 16];
    bv1.u[0] = *(const uint4*)&vT[((lane & 15) + 16) * 32 + base];
    bv1.u[1] = *(const uint4*)&vT[((lane & 15) + 16) * 32 + base + 16];
    acc0 = __builtin_amdgcn_wmma_f32_16x16x32_bf16(false, ap.v, false, bv0.v, (short)0, acc0, false, false);
    acc1 = __builtin_amdgcn_wmma_f32_16x16x32_bf16(false, ap.v, false, bv1.v, (short)0, acc1, false, false);
    __syncthreads();
  }

  // normalize and scatter O (only the DH=24 real features)
#pragma unroll
  for (int r = 0; r < 8; ++r) {
    float inv = 1.0f / l[r];
    int srow = qb * 64 + w * 16 + r + rhalf;
    float* op = O + (size_t)(b * S + srow) * D + h * DH;
    int f0 = lane & 15;            // 0..15 (always valid)
    int f1 = 16 + (lane & 15);     // valid only if < 24
    op[f0] = acc0[r] * inv;
    if (f1 < DH) op[f1] = acc1[r] * inv;
  }
}

// ---------------- launcher ----------------
extern "C" void kernel_launch(void* const* d_in, const int* in_sizes, int n_in,
                              void* d_out, int out_size, void* d_ws, size_t ws_size,
                              hipStream_t stream) {
  (void)in_sizes; (void)n_in; (void)out_size; (void)ws_size;
  const float* x  = (const float*)d_in[0];
  const unsigned char* mask = (const unsigned char*)d_in[1];
  const float* Wq = (const float*)d_in[2];
  const float* bq = (const float*)d_in[3];
  const float* Wk = (const float*)d_in[4];
  const float* bk = (const float*)d_in[5];
  const float* Wv = (const float*)d_in[6];
  const float* bv = (const float*)d_in[7];
  const float* Wo = (const float*)d_in[8];
  const float* bo = (const float*)d_in[9];

  char* ws = (char*)d_ws;
  __bf16* wb_qkv = (__bf16*)(ws + OFF_WB_QKV);
  float*  b_qkv  = (float*)(ws + OFF_B_QKV);
  __bf16* wb_o   = (__bf16*)(ws + OFF_WB_O);
  float*  b_o    = (float*)(ws + OFF_B_O);
  __bf16* qkv    = (__bf16*)(ws + OFF_QKV);
  float*  atto   = (float*)(ws + OFF_ATTO);

  // 1) weight prep (tiny)
  prep_qkv_w<<<(NT_QKV * KT * 32 * 16 + 255) / 256, 256, 0, stream>>>(Wq, bq, Wk, bk, Wv, bv, wb_qkv, b_qkv);
  prep_o_w<<<(NT_O * KT * 32 * 16 + 255) / 256, 256, 0, stream>>>(Wo, bo, wb_o, b_o);

  // 2) fused QKV projection: [32768 x 192] x [192 x 768] -> bf16 QKV (padded)
  gemm_wmma<1><<<dim3(MROWS / 64, NQKV / 64), 256, 0, stream>>>(x, wb_qkv, b_qkv, qkv, NQKV);

  // 3) flash attention: 64 query rows per block, one (b,h) each
  flash_attn<<<B * H * (S / 64), 128, 0, stream>>>(qkv, mask, atto);

  // 4) output projection: [32768 x 192] x [192 x 192] + bo -> d_out (f32)
  gemm_wmma<0><<<dim3(MROWS / 64, D / 64), 256, 0, stream>>>(atto, wb_o, b_o, (float*)d_out, D);
}
